// Compatability_11338713662137
// MI455X (gfx1250) — compile-verified, matrix operand order checked by reading, not run on hardware
//
#include <hip/hip_runtime.h>

typedef __attribute__((ext_vector_type(16))) _Float16 v16h;
typedef __attribute__((ext_vector_type(8)))  _Float16 v8h;
typedef __attribute__((ext_vector_type(8)))  float    v8f;

#define NB 48
#define SS 1024
#define DD 128
#define DK 64
#define TPB 1024  // (1024/32)^2 32x32 score tiles per batch

// log2(e) * 10  (exp(10*tanh(x)) == exp2(C*tanh(x)))
#define EXP_SCALE 14.426950408889634f

// CDNA5 hardware tanh (trans op). ISA 7.4.3: one independent op / V_NOP must
// follow a TRANS op before its result is consumed -> embed v_nop.
__device__ __forceinline__ float fast_tanh(float x) {
    float y;
    asm("v_tanh_f32 %0, %1\n\tv_nop" : "=v"(y) : "v"(x));
    return y;
}

// Build the documented 16-bit A/B operand (16x32 / 32x16) for one lane:
// two contiguous 8-half (16B) chunks at k-offsets hi*8 and 16+hi*8.
__device__ __forceinline__ v16h load16(const _Float16* row, int hi) {
    const v8h c0 = *(const v8h*)(row + hi * 8);
    const v8h c1 = *(const v8h*)(row + 16 + hi * 8);
    v16h r;
    #pragma unroll
    for (int e = 0; e < 8; ++e) { r[e] = c0[e]; r[8 + e] = c1[e]; }
    return r;
}

// ---------------------------------------------------------------------------
// Kernel 0: transpose+convert W [DD,DK] f32 -> Wt [DK,DD] f16, so the proj
// kernel's B operand loads are contiguous 16B chunks (no scalar gathers).
// ---------------------------------------------------------------------------
__global__ void prep_w_kernel(const float* __restrict__ Wq,
                              const float* __restrict__ Wk,
                              _Float16* __restrict__ Wqt,
                              _Float16* __restrict__ Wkt) {
    const int i = blockIdx.x * blockDim.x + threadIdx.x;  // DD*DK = 8192
    if (i < DD * DK) {
        const int d = i / DK, n = i % DK;
        Wqt[n * DD + d] = (_Float16)Wq[i];
        Wkt[n * DD + d] = (_Float16)Wk[i];
    }
}

// ---------------------------------------------------------------------------
// Kernel 1: projection. One wave -> 16(row) x 16(col) tile of BOTH Q and K.
// 3072 row tiles * 4 col tiles = 12288 waves, 8/block -> 1536 blocks.
// ---------------------------------------------------------------------------
__global__ __launch_bounds__(256) void proj_kernel(
    const float* __restrict__ query,
    const _Float16* __restrict__ Wqt,
    const _Float16* __restrict__ Wkt,
    _Float16* __restrict__ Qh,
    _Float16* __restrict__ Kh)
{
    const int lane    = threadIdx.x & 31;
    const int wave    = blockIdx.x * 8 + (threadIdx.x >> 5);
    const int colTile = wave & 3;
    const int rowTile = wave >> 2;
    const int row0    = rowTile * 16;
    const int n0      = colTile * 16;
    const int m       = lane & 15;
    const int hi      = lane >> 4;

    const float*    qrow = query + (size_t)(row0 + m) * DD;
    const _Float16* wq   = Wqt + (size_t)(n0 + m) * DD;
    const _Float16* wk   = Wkt + (size_t)(n0 + m) * DD;

    v8f accQ = {}, accK = {};
    #pragma unroll
    for (int ks = 0; ks < 4; ++ks) {             // D = 128 = 4 * 32
        const int kbase = ks * 32;
        const float* p0 = qrow + kbase + hi * 8;
        const float* p1 = qrow + kbase + 16 + hi * 8;
        v16h a;
        #pragma unroll
        for (int e = 0; e < 8; ++e) {
            a[e]     = (_Float16)p0[e];
            a[8 + e] = (_Float16)p1[e];
        }
        const v16h bq = load16(wq + kbase, hi);
        const v16h bk = load16(wk + kbase, hi);
        accQ = __builtin_amdgcn_wmma_f32_16x16x32_f16(
            false, a, false, bq, (short)0, accQ, false, false);
        accK = __builtin_amdgcn_wmma_f32_16x16x32_f16(
            false, a, false, bk, (short)0, accK, false, false);
    }
    #pragma unroll
    for (int r = 0; r < 8; ++r) {
        const size_t rr = (size_t)(row0 + r + hi * 8);
        Qh[rr * DK + n0 + m] = (_Float16)accQ[r];
        Kh[rr * DK + n0 + m] = (_Float16)accK[r];
    }
}

// ---------------------------------------------------------------------------
// Shared 32x32 score-tile computation. Pass 1 (WRITE=false) accumulates the
// per-tile exp-sum into partials[wave]; pass 2 (WRITE=true) recomputes and
// streams out the normalized result with non-temporal stores (written once,
// never re-read -> keep Q/K resident in L2 instead).
// ---------------------------------------------------------------------------
template <bool WRITE>
__device__ __forceinline__ void score_tile(
    const _Float16* __restrict__ Qh,
    const _Float16* __restrict__ Kh,
    float* __restrict__ out,
    const float* __restrict__ sums,
    float* __restrict__ partials)
{
    const int lane = threadIdx.x & 31;
    const int wave = blockIdx.x * 8 + (threadIdx.x >> 5);
    const int b    = wave >> 10;              // / TPB
    const int t    = wave & (TPB - 1);
    const int qt   = t >> 5;                  // 0..31
    const int st   = t & 31;                  // 0..31
    const int m    = lane & 15;
    const int hi   = lane >> 4;
    const int q0   = qt * 32;
    const int s0   = st * 32;

    const size_t base = (size_t)b * SS * DK;
    const _Float16* qr[2] = { Qh + base + (size_t)(q0 + m) * DK,
                              Qh + base + (size_t)(q0 + 16 + m) * DK };
    const _Float16* kr[2] = { Kh + base + (size_t)(s0 + m) * DK,
                              Kh + base + (size_t)(s0 + 16 + m) * DK };

    v8f acc[2][2] = {};
    #pragma unroll
    for (int ks = 0; ks < 2; ++ks) {          // DK = 64 = 2 * 32
        const int kb = ks * 32;
        v16h A[2], B[2];
        #pragma unroll
        for (int i = 0; i < 2; ++i) {
            A[i] = load16(qr[i] + kb, hi);
            B[i] = load16(kr[i] + kb, hi);
        }
        #pragma unroll
        for (int mi = 0; mi < 2; ++mi)
            #pragma unroll
            for (int ni = 0; ni < 2; ++ni)
                acc[mi][ni] = __builtin_amdgcn_wmma_f32_16x16x32_f16(
                    false, A[mi], false, B[ni], (short)0, acc[mi][ni],
                    false, false);
    }

    const float inv = WRITE ? (1.0f / sums[b]) : 0.0f;
    float* obase = out + (size_t)b * SS * SS;
    float lsum = 0.0f;

    #pragma unroll
    for (int mi = 0; mi < 2; ++mi)
        #pragma unroll
        for (int ni = 0; ni < 2; ++ni)
            #pragma unroll
            for (int r = 0; r < 8; ++r) {
                const int q = q0 + mi * 16 + r + hi * 8;
                const int s = s0 + ni * 16 + m;
                const float x = acc[mi][ni][r];
                // e = exp(10*tanh(x)); diagonal -> 0 (== -1e8 additive mask).
                // No max-pass needed: 10*tanh(x) in [-10,10], exp safe in f32.
                const float e = (q == s)
                    ? 0.0f
                    : __builtin_amdgcn_exp2f(EXP_SCALE * fast_tanh(x));
                if (WRITE) {
                    __builtin_nontemporal_store(e * inv,
                                                obase + (size_t)q * SS + s);
                } else {
                    lsum += e;
                }
            }

    if (!WRITE) {
        #pragma unroll
        for (int off = 16; off > 0; off >>= 1)
            lsum += __shfl_down(lsum, off, 32);
        if (lane == 0) partials[wave] = lsum;   // deterministic: no atomics
    }
}

__global__ __launch_bounds__(256) void sum_pass_kernel(
    const _Float16* __restrict__ Qh, const _Float16* __restrict__ Kh,
    float* __restrict__ partials)
{
    score_tile<false>(Qh, Kh, nullptr, nullptr, partials);
}

__global__ __launch_bounds__(256) void write_pass_kernel(
    const _Float16* __restrict__ Qh, const _Float16* __restrict__ Kh,
    const float* __restrict__ sums, float* __restrict__ out)
{
    score_tile<true>(Qh, Kh, out, sums, nullptr);
}

// ---------------------------------------------------------------------------
// Kernel 3: deterministic fixed-order reduction of per-tile partials.
// One block per batch, 256 threads, TPB=1024 partials each.
// ---------------------------------------------------------------------------
__global__ __launch_bounds__(256) void reduce_kernel(
    const float* __restrict__ partials, float* __restrict__ sums)
{
    __shared__ float sm[256];
    const int b = blockIdx.x;
    float s = 0.0f;
    for (int t = threadIdx.x; t < TPB; t += 256)
        s += partials[(size_t)b * TPB + t];
    sm[threadIdx.x] = s;
    __syncthreads();
    #pragma unroll
    for (int off = 128; off > 0; off >>= 1) {
        if (threadIdx.x < off) sm[threadIdx.x] += sm[threadIdx.x + off];
        __syncthreads();
    }
    if (threadIdx.x == 0) sums[b] = sm[0];
}

// ---------------------------------------------------------------------------
// Launch
// ---------------------------------------------------------------------------
extern "C" void kernel_launch(void* const* d_in, const int* in_sizes, int n_in,
                              void* d_out, int out_size, void* d_ws, size_t ws_size,
                              hipStream_t stream) {
    const float* query = (const float*)d_in[0];
    // d_in[1] (exchange) and d_in[2] (solution_indexes) unused by reference
    const float* Wq = (const float*)d_in[3];
    const float* Wk = (const float*)d_in[4];
    float* out = (float*)d_out;

    _Float16* Qh  = (_Float16*)d_ws;                       // 6 MB
    _Float16* Kh  = Qh + (size_t)NB * SS * DK;             // 6 MB
    _Float16* Wqt = Kh + (size_t)NB * SS * DK;             // 16 KB
    _Float16* Wkt = Wqt + (size_t)DK * DD;                 // 16 KB
    float* partials = (float*)(Wkt + (size_t)DK * DD);     // 192 KB
    float* sums     = partials + (size_t)NB * TPB;         // 192 B

    prep_w_kernel<<<32, 256, 0, stream>>>(Wq, Wk, Wqt, Wkt);
    proj_kernel<<<1536, 256, 0, stream>>>(query, Wqt, Wkt, Qh, Kh);

    // 48 batches * 1024 tiles = 49152 waves, 8/block -> 6144 blocks
    sum_pass_kernel<<<6144, 256, 0, stream>>>(Qh, Kh, partials);
    reduce_kernel<<<NB, 256, 0, stream>>>(partials, sums);
    write_pass_kernel<<<6144, 256, 0, stream>>>(Qh, Kh, sums, out);
}
